// Discriminators_2l_76081050681687
// MI455X (gfx1250) — matrix operand order repaired and checked by reading, not compiled, not run on hardware
//
#include <hip/hip_runtime.h>

#define BATCH 2048
#define FEAT 512
#define HIDDEN 512
#define TPB 512
#define WAVES 16
#define TILES 2   // 16-row WMMA tiles per wave (16 waves * 32 rows = 512 hidden)

typedef __attribute__((ext_vector_type(16))) _Float16 v16h;
typedef __attribute__((ext_vector_type(8)))  _Float16 v8h;
typedef __attribute__((ext_vector_type(8)))  float    v8f;

union AFrag { v16h v; v8h h[2]; };

__global__ __launch_bounds__(TPB)
void disc2l_kernel(const float* __restrict__ Z,
                   const int*   __restrict__ y,
                   const float* __restrict__ W1,
                   const float* __restrict__ b1,
                   const float* __restrict__ W2,
                   const float* __restrict__ b2,
                   float*       __restrict__ out)
{
    __shared__ alignas(32) _Float16 zsh[FEAT];   // Z[b] as f16: B-matrix source
    __shared__ float partial[2 * WAVES];

    const int b    = blockIdx.x;
    const int tid  = threadIdx.x;
    const int lane = tid & 31;       // wave32
    const int wave = tid >> 5;       // 16 waves
    const int cls  = y[b];

    // Stage Z[b] (f32 -> f16) into LDS once.
    zsh[tid] = (_Float16)Z[(size_t)b * FEAT + tid];
    __syncthreads();

    const int half = lane >> 4;      // 0: lanes 0-15, 1: lanes 16-31
    const int m    = lane & 15;      // row within a 16-row tile
    const int cb   = half << 3;      // A-fragment K sub-offset: 0 or 8

    // Per-lane base row pointer; tile offsets fold into load immediates.
    const float* base = W1 + (size_t)cls * (size_t)(HIDDEN * FEAT)
                           + (size_t)(wave * 32 + m) * FEAT + cb;

    v8f buf0[2 * TILES], buf1[2 * TILES];
    v8f acc[TILES] = {};

    auto loadk = [&](v8f* d, int k) {
#pragma unroll
        for (int t = 0; t < TILES; ++t) {
            d[2 * t]     = *(const v8f*)(base + t * 16 * FEAT + k);
            d[2 * t + 1] = *(const v8f*)(base + t * 16 * FEAT + k + 16);
        }
    };
    auto compute = [&](const v8f* d, int k) {
        // B fragment: 32x16 f16, column-replicated Z chunk (two 16 B LDS reads at k+8h, k+16+8h).
        AFrag bfu;
        bfu.h[0] = *(const v8h*)&zsh[k + cb];
        bfu.h[1] = *(const v8h*)&zsh[k + 16 + cb];
        const v16h bf = bfu.v;
#pragma unroll
        for (int t = 0; t < TILES; ++t) {
            AFrag a;
            a.h[0] = __builtin_convertvector(d[2 * t],     v8h);
            a.h[1] = __builtin_convertvector(d[2 * t + 1], v8h);
            acc[t] = __builtin_amdgcn_wmma_f32_16x16x32_f16(
                false, a.v, false, bf, (short)0, acc[t], false, false);
        }
    };

    // Ping-pong double-buffered K loop: loads for the next 32-chunk are in
    // flight while the current chunk converts + WMMAs.
    loadk(buf0, 0);
    for (int k0 = 0; k0 < FEAT; k0 += 64) {
        loadk(buf1, k0 + 32);
        if (k0 + 128 < FEAT) {         // uniform guard, once per iteration
#pragma unroll
            for (int t = 0; t < TILES; ++t) {
                __builtin_prefetch(base + t * 16 * FEAT + k0 + 128, 0, 3);
                __builtin_prefetch(base + t * 16 * FEAT + k0 + 160, 0, 3);
            }
        }
        compute(buf0, k0);
        const int kn = (k0 + 64 < FEAT) ? (k0 + 64) : (k0 + 32); // tail: cached reload
        loadk(buf0, kn);
        compute(buf1, k0 + 32);
    }

    // Epilogue: D columns are identical; lane's VGPR r holds tile row r + 8*half.
    // Only n==0 lanes (2 per wave) contribute. tanh via exp2/rcp (saturation-safe).
    if (m == 0) {
        const size_t rb = (size_t)cls * HIDDEN + (size_t)(wave * 32 + half * 8);
        float s = 0.0f;
#pragma unroll
        for (int t = 0; t < TILES; ++t) {
            const v8f bv = *(const v8f*)(b1 + rb + t * 16);
            const v8f wv = *(const v8f*)(W2 + rb + t * 16);
#pragma unroll
            for (int r = 0; r < 8; ++r) {
                const float x  = acc[t][r] + bv[r];
                const float e  = __builtin_amdgcn_exp2f(x * 2.885390081777927f); // 2*log2(e)
                const float th = 1.0f - 2.0f * __builtin_amdgcn_rcpf(e + 1.0f);
                s += th * wv[r];
            }
        }
        partial[wave * 2 + half] = s;
    }
    __syncthreads();

    if (tid == 0) {
        float s = 0.0f;
#pragma unroll
        for (int i = 0; i < 2 * WAVES; ++i) s += partial[i];
        out[b] = s + b2[cls];
    }
}

extern "C" void kernel_launch(void* const* d_in, const int* in_sizes, int n_in,
                              void* d_out, int out_size, void* d_ws, size_t ws_size,
                              hipStream_t stream) {
    (void)in_sizes; (void)n_in; (void)out_size; (void)d_ws; (void)ws_size;
    const float* Z  = (const float*)d_in[0];
    const int*   y  = (const int*)  d_in[1];
    const float* W1 = (const float*)d_in[2];
    const float* b1 = (const float*)d_in[3];
    const float* W2 = (const float*)d_in[4];
    const float* b2 = (const float*)d_in[5];
    float* out = (float*)d_out;
    disc2l_kernel<<<BATCH, TPB, 0, stream>>>(Z, y, W1, b1, W2, b2, out);
}
